// Conv2d_cd_pixel_difference_matrix5x5_shared_62002147885524
// MI455X (gfx1250) — matrix operand order repaired and checked by reading, not compiled, not run on hardware
//
#include <hip/hip_runtime.h>
#include <hip/hip_bf16.h>

typedef __attribute__((ext_vector_type(16))) __bf16     v16bf;
typedef __attribute__((ext_vector_type(8)))  float      v8f;
typedef __attribute__((ext_vector_type(8)))  unsigned   v8u;
typedef __attribute__((ext_vector_type(4)))  unsigned   v4u;
typedef unsigned int u32;

#define CIN   64
#define COUT  64
#define HW    128
#define KSTEPS 18              // 576 / 32
#define MTILES 4               // COUT / 16
#define CHUNK_C 32             // channels per LDS chunk
#define NPAIR  16              // channel pairs per chunk
#define CHUNK_KSTEPS 9         // one k-step per 3x3 tap
#define CC_N   130             // cols: x=-1..128 (zero-padded ends)
#define P_STRIDE 20            // 16 pairs padded to 20 words: 80B = 16B-aligned,
                               // gcd(20,64)=4 -> conflict-free b128 banking
#define THETA 0.7f

// round-to-nearest-even fp32 -> bf16 bits
__device__ __forceinline__ u32 bf16_rne(float f) {
    u32 u = __float_as_uint(f);
    u32 r = (u + 0x7FFFu + ((u >> 16) & 1u)) >> 16;
    return r & 0xFFFFu;
}
__device__ __forceinline__ float bf16_to_f(u32 b) { return __uint_as_float(b << 16); }

// load 8 consecutive u32 from LDS as two aligned b128s, concat to v8u
__device__ __forceinline__ v8u lds_v8(const u32* p) {
    v4u a = *(const v4u*)p;
    v4u b = *(const v4u*)(p + 4);
    return __builtin_shufflevector(a, b, 0, 1, 2, 3, 4, 5, 6, 7);
}

// ---------------------------------------------------------------------------
// K ordering (our choice, consistent across A and B):
//   global k-step t = chunk*9 + tap, tap = ky*3+kx;  K-within-step = channel
//   (chunk c covers channels c*32 .. c*32+31)
//
// Stage 1: W_eff[o][i][tap] = W - theta*rowsum at center tap (tap==4),
// split bf16 hi/lo, packed for V_WMMA_F32_16X16X32_BF16 A in contiguous k order (same as B):
//   lanes 0-15 : M=lane,   V0..V7 -> kk=0..15
//   lanes 16-31: M=lane-16,V0..V7 -> kk=16..31
// stored as packed u32 pairs (kk even in [15:0], kk+1 in [31:16]).
// ---------------------------------------------------------------------------
__global__ __launch_bounds__(256) void prep_weights(
        const float* __restrict__ W, u32* __restrict__ Ahi, u32* __restrict__ Alo) {
    int t = blockIdx.x * 256 + threadIdx.x;          // 0 .. 4*18*32-1
    if (t >= MTILES * KSTEPS * 32) return;
    int lane   = t & 31;
    int rest   = t >> 5;
    int kstep  = rest % KSTEPS;                      // global k-step
    int mt     = rest / KSTEPS;
    int M      = mt * 16 + (lane & 15);
    int laneHi = lane >> 4;
    int chunk  = kstep / CHUNK_KSTEPS;
    int tap    = kstep - chunk * CHUNK_KSTEPS;       // 0..8 = ky*3+kx

    u32 base = ((u32)(mt * KSTEPS + kstep) * 32 + lane) * 8;
#pragma unroll
    for (int v = 0; v < 8; ++v) {
        int kk = 2 * v + (laneHi ? 16 : 0); // A-layout K
        u32 ph = 0, pl = 0;
#pragma unroll
        for (int e = 0; e < 2; ++e) {
            int i = chunk * CHUNK_C + kk + e;        // input channel
            const float* wrow = W + ((size_t)M * CIN + i) * 9;
            float w = wrow[tap];
            if (tap == 4) {
                float ssum = 0.f;
#pragma unroll
                for (int jj = 0; jj < 9; ++jj) ssum += wrow[jj];
                w -= THETA * ssum;
            }
            u32 h = bf16_rne(w);
            u32 l = bf16_rne(w - bf16_to_f(h));
            ph |= h << (16 * e);
            pl |= l << (16 * e);
        }
        Ahi[base + v] = ph;
        Alo[base + v] = pl;
    }
}

// ---------------------------------------------------------------------------
// Stage 2: one workgroup (8 waves) per (batch n, output row y).
// Phase A: for channel pair p and s-row r, at column cc (= x+1):
//   Sh[r][cc][p] = bf16_hi(s[2p]) | bf16_hi(s[2p+1])<<16
//   Sl[r][cc][p] = bf16_lo(s[2p]) | bf16_lo(s[2p+1])<<16
//   where s = box3x3_reflect(x)[ch][y-1+r][cc-1]; rows with y-1+r outside
//   the image stay zero (conv zero padding), cc=0/129 are zero columns.
// Phase B: implicit GEMM out[64x128] = W_eff[64x576] @ P[576x128] via
//   v_wmma_f32_16x16x32_bf16, hi/lo split (3 WMMA per tile). With pairs
//   innermost, each B operand is 8 consecutive words -> two ds_load_b128
//   straight into the operand registers: zero repack VALU in the hot loop.
// ---------------------------------------------------------------------------
__global__ __launch_bounds__(256) void conv_row(
        const float* __restrict__ x,
        const u32* __restrict__ Ahi, const u32* __restrict__ Alo,
        float* __restrict__ out) {
    __shared__ __align__(16) u32 Sh[3 * CC_N * P_STRIDE];   // 31200 B
    __shared__ __align__(16) u32 Sl[3 * CC_N * P_STRIDE];   // 31200 B

    const int y   = blockIdx.x;
    const int n   = blockIdx.y;
    const int tid = threadIdx.x;
    const int lane   = tid & 31;
    const int wv     = tid >> 5;          // N-tile id, 0..7
    const int laneN  = lane & 15;
    const int laneHi = lane >> 4;
    const int N      = wv * 16 + laneN;   // output pixel x

    const float* xn = x + (size_t)n * CIN * HW * HW;
    const v8u* A8hi = (const v8u*)Ahi;
    const v8u* A8lo = (const v8u*)Alo;

    v8f acc[MTILES];
#pragma unroll
    for (int m = 0; m < MTILES; ++m) acc[m] = (v8f){};

    // zero both planes once (pad cols/words + out-of-range rows stay zero)
    for (int f = tid; f < 3 * CC_N * P_STRIDE; f += 256) { Sh[f] = 0u; Sl[f] = 0u; }
    __syncthreads();

    // per-lane bases: column N, this half-wave's 8 pairs
    const u32* ShB = Sh + N * P_STRIDE + laneHi * 8;
    const u32* SlB = Sl + N * P_STRIDE + laneHi * 8;

    for (int chunk = 0; chunk < 2; ++chunk) {
        // ---- fill S planes: 2 channels (one pair) per thread-iteration ----
        for (int f = tid; f < NPAIR * 3 * HW; f += 256) {  // 6144
            int xi    = f & 127;
            int rowid = f >> 7;               // 0..47
            int p     = rowid & (NPAIR - 1);
            int r     = rowid >> 4;           // 0..2 (s row y-1+r)
            int yp    = y - 1 + r;
            if (yp >= 0 && yp <= 127) {
                const float* xc0 = xn + (size_t)(chunk * CHUNK_C + 2 * p) * HW * HW;
                int xl = xi - 1; if (xl < 0)   xl = 1;     // reflect(-1)=1
                int xr = xi + 1; if (xr > 127) xr = 126;   // reflect(128)=126
                float s0 = 0.f, s1 = 0.f;
#pragma unroll
                for (int dy = -1; dy <= 1; ++dy) {
                    int ry = yp + dy;
                    ry = ry < 0 ? -ry : (ry > 127 ? 254 - ry : ry);
                    const float* row0 = xc0 + ry * HW;
                    const float* row1 = row0 + HW * HW;    // channel 2p+1
                    s0 += row0[xl] + row0[xi] + row0[xr];
                    s1 += row1[xl] + row1[xi] + row1[xr];
                }
                u32 h0 = bf16_rne(s0), h1 = bf16_rne(s1);
                u32 l0 = bf16_rne(s0 - bf16_to_f(h0));
                u32 l1 = bf16_rne(s1 - bf16_to_f(h1));
                int idx = (r * CC_N + xi + 1) * P_STRIDE + p;
                Sh[idx] = h0 | (h1 << 16);
                Sl[idx] = l0 | (l1 << 16);
            }
        }
        __syncthreads();

        // ---- GEMM over this chunk's 9 k-steps (taps) ----
#pragma unroll
        for (int tt = 0; tt < CHUNK_KSTEPS; ++tt) {
            const int ky  = tt / 3;                    // compile-time after unroll
            const int kx  = tt - 3 * ky;
            const int off = (ky * CC_N + kx) * P_STRIDE;  // immediate offset

            // B operand: 8 consecutive pair-words -> 2x ds_load_b128 each
            v8u bh = lds_v8(ShB + off);
            v8u bl = lds_v8(SlB + off);
            v16bf Bh = __builtin_bit_cast(v16bf, bh);
            v16bf Bl = __builtin_bit_cast(v16bf, bl);

            const int t = chunk * CHUNK_KSTEPS + tt;
#pragma unroll
            for (int m = 0; m < MTILES; ++m) {
                v8u  ah  = A8hi[(m * KSTEPS + t) * 32 + lane];
                v8u  al  = A8lo[(m * KSTEPS + t) * 32 + lane];
                v16bf Ah = __builtin_bit_cast(v16bf, ah);
                v16bf Al = __builtin_bit_cast(v16bf, al);
                // split-precision accumulate: hi*hi + hi*lo + lo*hi (f32 C)
                acc[m] = __builtin_amdgcn_wmma_f32_16x16x32_bf16(
                             false, Ah, false, Bh, (short)0, acc[m], false, false);
                acc[m] = __builtin_amdgcn_wmma_f32_16x16x32_bf16(
                             false, Ah, false, Bl, (short)0, acc[m], false, false);
                acc[m] = __builtin_amdgcn_wmma_f32_16x16x32_bf16(
                             false, Al, false, Bh, (short)0, acc[m], false, false);
            }
        }
        __syncthreads();   // S reused by next chunk
    }

    // ---- epilogue: D layout = VGPR r -> M=r (lanes 0-15) / M=r+8 (16-31) ----
#pragma unroll
    for (int m = 0; m < MTILES; ++m) {
#pragma unroll
        for (int r = 0; r < 8; ++r) {
            int o = m * 16 + laneHi * 8 + r;
            out[(((size_t)n * COUT + o) * HW + y) * HW + N] = acc[m][r];
        }
    }
}

extern "C" void kernel_launch(void* const* d_in, const int* in_sizes, int n_in,
                              void* d_out, int out_size, void* d_ws, size_t ws_size,
                              hipStream_t stream) {
    const float* x = (const float*)d_in[0];   // [16,64,128,128]
    const float* W = (const float*)d_in[1];   // [64,64,3,3]
    float* out = (float*)d_out;               // [16,64,128,128]

    const size_t a_elems = (size_t)MTILES * KSTEPS * 32 * 8;   // u32 each
    if (ws_size < 2 * a_elems * sizeof(u32)) return;           // need ~148 KB
    u32* Ahi = (u32*)d_ws;
    u32* Alo = Ahi + a_elems;

    prep_weights<<<9, 256, 0, stream>>>(W, Ahi, Alo);
    conv_row<<<dim3(HW, 16), 256, 0, stream>>>(x, Ahi, Alo, out);
}
